// DeformableAttnBlock_47287589929861
// MI455X (gfx1250) — matrix operand order repaired and checked by reading, not compiled, not run on hardware
//
#include <hip/hip_runtime.h>
#include <hip/hip_bf16.h>

typedef __attribute__((ext_vector_type(16))) _Float16 v16h;
typedef __attribute__((ext_vector_type(8)))  _Float16 v8h;
typedef __attribute__((ext_vector_type(4)))  _Float16 v4h;
typedef __attribute__((ext_vector_type(8)))  float    v8f;

#define LRELU_SLOPE 0.1f

// Problem geometry (fixed by the reference)
constexpr int Hc  = 192;
constexpr int Wc  = 192;
constexpr int HWc = Hc * Wc;          // 36864
constexpr int PADc = 2;               // covers dil=1 and dil=2 convs
constexpr int Hp  = Hc + 2 * PADc;    // 196
constexpr int Wp  = Wc + 2 * PADc;    // 196
constexpr long HALO = (long)(Hp * Wp - Hc * Wc); // 1552 halo pixels / image

__device__ __forceinline__ v16h cat8(v8h lo, v8h hi) {
  return __builtin_shufflevector(lo, hi, 0,1,2,3,4,5,6,7,8,9,10,11,12,13,14,15);
}
__device__ __forceinline__ int iclampi(int v, int lo, int hi) {
  return v < lo ? lo : (v > hi ? hi : v);
}

// ---------------------------------------------------------------------------
// Zero the 2-pixel halo of a padded channel-last f16 buffer [nimg][Hp][Wp][cp]
// ---------------------------------------------------------------------------
__global__ void zero_halo_kernel(_Float16* __restrict__ buf, int nimg, int cp)
{
  long idx = (long)blockIdx.x * blockDim.x + threadIdx.x;
  const long total = (long)nimg * HALO * cp;
  if (idx >= total) return;
  const int c = (int)(idx % cp);
  long t = idx / cp;
  const int hi = (int)(t % HALO);
  const int img = (int)(t / HALO);
  int y, x;
  if (hi < 2 * Wp)            { y = hi / Wp;               x = hi % Wp; }
  else if (hi < 4 * Wp)       { int i = hi - 2 * Wp; y = (Hp - 2) + i / Wp; x = i % Wp; }
  else                        { int i = hi - 4 * Wp; y = 2 + i / 4;
                                int xi = i % 4; x = (xi < 2) ? xi : (Wp - 4 + xi); }
  buf[(((long)img * Hp + y) * Wp + x) * cp + c] = (_Float16)0.0f;
}

// ---------------------------------------------------------------------------
// Weight prep: conv f32 (cout,cin,3,3) -> f16 [tap][cout][cinp], zero-padded K
// ---------------------------------------------------------------------------
__global__ void convprep_kernel(const float* __restrict__ w, _Float16* __restrict__ o,
                                int cout, int cin, int cinp)
{
  long idx = (long)blockIdx.x * blockDim.x + threadIdx.x;
  const long total = (long)9 * cout * cinp;
  if (idx >= total) return;
  const int k  = (int)(idx % cinp);
  long t = idx / cinp;
  const int co = (int)(t % cout);
  const int tap = (int)(t / cout);
  const int ky = tap / 3, kx = tap % 3;
  float v = 0.0f;
  if (k < cin) v = w[(((long)co * cin + k) * 3 + ky) * 3 + kx];
  o[idx] = (_Float16)v;
}

// GEMM weight prep: f32 (K=64, N) row-major -> f16 transposed [n][k]
__global__ void gemmprep_kernel(const float* __restrict__ w, _Float16* __restrict__ o, int N)
{
  long idx = (long)blockIdx.x * blockDim.x + threadIdx.x;
  if (idx >= (long)N * 64) return;
  const int k = (int)(idx % 64);
  const int n = (int)(idx / 64);
  o[idx] = (_Float16)w[(long)k * N + n];
}

// ---------------------------------------------------------------------------
// f32 channel-first (nimg,C,H,W) -> f16 padded channel-last [img][Hp][Wp][cp]+cbase
// ---------------------------------------------------------------------------
__global__ void cvt_cf2cl_kernel(const float* __restrict__ in, _Float16* __restrict__ o,
                                 int nimg, int C, int cp, int cbase)
{
  long idx = (long)blockIdx.x * blockDim.x + threadIdx.x;
  const long total = (long)nimg * C * HWc;
  if (idx >= total) return;
  const int c = (int)(idx % C);
  long t = idx / C;
  const int p = (int)(t % HWc);
  const int img = (int)(t / HWc);
  const int y = p / Wc, x = p % Wc;
  o[(((long)img * Hp + y + PADc) * Wp + (x + PADc)) * cp + cbase + c] =
      (_Float16)in[((long)img * C + c) * HWc + p];
}

// ---------------------------------------------------------------------------
// Warp (border bilinear) + concat -> qk_in, padded channel-last f16, cp=224
// channels: [warp01(64)|frame1(64)|warp21(64)|flow_f(4)|flow_b(4)|zeros(24)]
// ---------------------------------------------------------------------------
__global__ void warp_concat_kernel(const float* __restrict__ frame,
                                   const float* __restrict__ flowf,
                                   const float* __restrict__ flowb,
                                   _Float16* __restrict__ qk16, int B)
{
  long idx = (long)blockIdx.x * blockDim.x + threadIdx.x;
  if (idx >= (long)B * HWc) return;
  const int b = (int)(idx / HWc), p = (int)(idx % HWc);
  const int y = p / Wc, x = p % Wc;
  _Float16* qb = qk16 + (((long)b * Hp + y + PADc) * Wp + (x + PADc)) * 224;

  const float* fb0 = flowb + (size_t)b * 4 * HWc;
  const float* ff1 = flowf + ((size_t)b * 4 + 2) * HWc;

  float px0 = fminf(fmaxf((float)x + fb0[p],        0.0f), (float)(Wc - 1));
  float py0 = fminf(fmaxf((float)y + fb0[HWc + p],  0.0f), (float)(Hc - 1));
  float px2 = fminf(fmaxf((float)x + ff1[p],        0.0f), (float)(Wc - 1));
  float py2 = fminf(fmaxf((float)y + ff1[HWc + p],  0.0f), (float)(Hc - 1));

  const int x00 = (int)floorf(px0), y00 = (int)floorf(py0);
  const float wx0 = px0 - (float)x00, wy0 = py0 - (float)y00;
  const int x01 = iclampi(x00 + 1, 0, Wc - 1), y01 = iclampi(y00 + 1, 0, Hc - 1);
  const int x20 = (int)floorf(px2), y20 = (int)floorf(py2);
  const float wx2 = px2 - (float)x20, wy2 = py2 - (float)y20;
  const int x21 = iclampi(x20 + 1, 0, Wc - 1), y21 = iclampi(y20 + 1, 0, Hc - 1);

  const float* f0 = frame + (size_t)b * 3 * 64 * HWc;
  const float* f1 = f0 + (size_t)64 * HWc;
  const float* f2 = f1 + (size_t)64 * HWc;

  for (int c = 0; c < 64; ++c) {
    const float* i0 = f0 + (size_t)c * HWc;
    float v00 = i0[y00 * Wc + x00], v10 = i0[y00 * Wc + x01];
    float v01 = i0[y01 * Wc + x00], v11 = i0[y01 * Wc + x01];
    float w01v = (v00 * (1.0f - wx0) + v10 * wx0) * (1.0f - wy0)
               + (v01 * (1.0f - wx0) + v11 * wx0) * wy0;
    const float* i2 = f2 + (size_t)c * HWc;
    float u00 = i2[y20 * Wc + x20], u10 = i2[y20 * Wc + x21];
    float u01 = i2[y21 * Wc + x20], u11 = i2[y21 * Wc + x21];
    float w21v = (u00 * (1.0f - wx2) + u10 * wx2) * (1.0f - wy2)
               + (u01 * (1.0f - wx2) + u11 * wx2) * wy2;
    qb[c]       = (_Float16)w01v;
    qb[64 + c]  = (_Float16)f1[(size_t)c * HWc + p];
    qb[128 + c] = (_Float16)w21v;
  }
  for (int j = 0; j < 4; ++j) {
    qb[192 + j] = (_Float16)flowf[((size_t)b * 4 + j) * HWc + p];
    qb[196 + j] = (_Float16)flowb[((size_t)b * 4 + j) * HWc + p];
  }
  for (int j = 200; j < 224; ++j) qb[j] = (_Float16)0.0f;
}

// ---------------------------------------------------------------------------
// 3x3 conv, WMMA implicit GEMM, fully branchless inner loop:
//   A = f16 weights [tap][cout][cinp] (K contiguous)  -> 2x b128 loads
//   B = f16 padded channel-last activations           -> 2x b128 loads
// One wave = 16(cout) x 16(pixels along x) tile. Wave-uniform guard only.
// Epilogue: bias / leaky / residual; optional f32 chan-first + f16 chan-last out.
// ---------------------------------------------------------------------------
__global__ void __launch_bounds__(128)
conv3x3_wmma_kernel(const _Float16* __restrict__ x16, int cinp,
                    const _Float16* __restrict__ w16,
                    const float* __restrict__ bias,
                    const float* __restrict__ resid,
                    float* __restrict__ out32,
                    _Float16* __restrict__ out16, int o16pitch, int o16base,
                    int nimg, int cout, int dil, int leaky)
{
  const int lane = threadIdx.x & 31;
  const int wid  = threadIdx.x >> 5;
  const int xTiles = Wc >> 4, pixTiles = HWc >> 4, coTiles = cout >> 4;
  long tile = (long)blockIdx.x * 4 + wid;              // wave-uniform
  if (tile >= (long)nimg * coTiles * pixTiles) return;

  const int  pt  = (int)(tile % pixTiles);
  long r1 = tile / pixTiles;
  const int  cot = (int)(r1 % coTiles);
  const int  img = (int)(r1 / coTiles);
  const int  y   = pt / xTiles;
  const int  x0  = (pt % xTiles) << 4;
  const int  co0 = cot << 4;

  const int aM = lane & 15, bN = lane & 15;
  const int aKhi = (lane & 16) ? 16 : 0;
  const int bKhi = (lane & 16) ? 16 : 0;

  const _Float16* xb  = x16 + ((long)img * Hp * Wp) * cinp;
  const _Float16* ap0 = w16 + (long)(co0 + aM) * cinp + aKhi;
  const long tapStride = (long)cout * cinp;

  v8f acc = {};
  for (int ky = 0; ky < 3; ++ky) {
    const int yy = y + PADc + (ky - 1) * dil;
    #pragma unroll
    for (int kx = 0; kx < 3; ++kx) {
      const int xs = x0 + bN + PADc + (kx - 1) * dil;
      const _Float16* bp = xb + ((long)yy * Wp + xs) * cinp + bKhi;
      const _Float16* ap = ap0 + (long)(ky * 3 + kx) * tapStride;
      for (int k0 = 0; k0 < cinp; k0 += 32) {
        v8h a0 = *(const v8h*)(ap + k0);
        v8h a1 = *(const v8h*)(ap + k0 + 8);
        v8h b0 = *(const v8h*)(bp + k0);
        v8h b1 = *(const v8h*)(bp + k0 + 8);
        acc = __builtin_amdgcn_wmma_f32_16x16x32_f16(false, cat8(a0, a1), false,
                                                     cat8(b0, b1), (short)0, acc,
                                                     false, false);
      }
    }
  }

  const int n = lane & 15, mhi = (lane & 16) ? 8 : 0;
  float vv[8];
  #pragma unroll
  for (int e = 0; e < 8; ++e) {
    const int co = co0 + e + mhi;
    float v = acc[e] + (bias ? bias[co] : 0.0f);
    if (leaky) v = (v >= 0.0f) ? v : LRELU_SLOPE * v;
    if (resid) v += resid[((long)img * cout + co) * HWc + (long)y * Wc + (x0 + n)];
    vv[e] = v;
  }
  if (out32) {
    #pragma unroll
    for (int e = 0; e < 8; ++e)
      out32[((long)img * cout + co0 + e + mhi) * HWc + (long)y * Wc + (x0 + n)] = vv[e];
  }
  if (out16) {
    v8h o;
    #pragma unroll
    for (int e = 0; e < 8; ++e) o[e] = (_Float16)vv[e];
    *(v8h*)(out16 + (((long)img * Hp + y + PADc) * Wp + (x0 + n + PADc)) * o16pitch
            + o16base + co0 + mhi) = o;
  }
}

// ---------------------------------------------------------------------------
// GEMM over rows r=(b,tt,p), K=64, via WMMA:
//   A = transposed f16 weights [n][64] (M = out-channel)  -> 2x b128
//   B = f16 padded channel-last activations, cbase=tt*64  -> 2x b128
// Epilogue: optional strided f32 out, optional f16 padded chan-last out.
// ---------------------------------------------------------------------------
__global__ void __launch_bounds__(128)
gemm_wmma_kernel(const _Float16* __restrict__ x16, int xpitch,
                 const _Float16* __restrict__ wT,
                 const float* __restrict__ bias,
                 float* __restrict__ out32, long ob, long ot, long op, long on,
                 _Float16* __restrict__ out16, int o16pitch,
                 int B, int N, int leaky)
{
  const int T = 3;
  const int lane = threadIdx.x & 31;
  const int wid  = threadIdx.x >> 5;
  const int nTiles = N >> 4, pTiles = HWc >> 4;
  long tile = (long)blockIdx.x * 4 + wid;              // wave-uniform
  if (tile >= (long)B * T * pTiles * nTiles) return;

  const int nt = (int)(tile % nTiles);
  long r1 = tile / nTiles;
  const int pt = (int)(r1 % pTiles);
  r1 /= pTiles;
  const int tt = (int)(r1 % T);
  const int bb = (int)(r1 / T);
  const int p0 = pt << 4, n0 = nt << 4;
  const int y = p0 / Wc, xbase = p0 % Wc;

  const int aM = lane & 15, bN = lane & 15;
  const int aKhi = (lane & 16) ? 16 : 0;
  const int bKhi = (lane & 16) ? 16 : 0;

  const _Float16* ap = wT + (long)(n0 + aM) * 64 + aKhi;
  const _Float16* bp = x16 + (((long)bb * Hp + y + PADc) * Wp + (xbase + bN + PADc)) * xpitch
                     + tt * 64 + bKhi;

  v8f acc = {};
  #pragma unroll
  for (int k0 = 0; k0 < 64; k0 += 32) {
    v8h a0 = *(const v8h*)(ap + k0);
    v8h a1 = *(const v8h*)(ap + k0 + 8);
    v8h b0 = *(const v8h*)(bp + k0);
    v8h b1 = *(const v8h*)(bp + k0 + 8);
    acc = __builtin_amdgcn_wmma_f32_16x16x32_f16(false, cat8(a0, a1), false,
                                                 cat8(b0, b1), (short)0, acc,
                                                 false, false);
  }

  const int mhi = (lane & 16) ? 8 : 0;
  float vv[8];
  #pragma unroll
  for (int e = 0; e < 8; ++e) {
    const int n = n0 + e + mhi;
    float v = acc[e] + (bias ? bias[n] : 0.0f);
    if (leaky) v = (v >= 0.0f) ? v : LRELU_SLOPE * v;
    vv[e] = v;
  }
  if (out32) {
    const long base = (long)bb * ob + (long)tt * ot + (long)(p0 + bN) * op;
    #pragma unroll
    for (int e = 0; e < 8; ++e)
      out32[base + (long)(n0 + e + mhi) * on] = vv[e];
  }
  if (out16) {
    v8h o;
    #pragma unroll
    for (int e = 0; e < 8; ++e) o[e] = (_Float16)vv[e];
    const int img = bb * T + tt;
    *(v8h*)(out16 + (((long)img * Hp + y + PADc) * Wp + (xbase + bN + PADc)) * o16pitch
            + n0 + mhi) = o;
  }
}

// ---------------------------------------------------------------------------
// Softmax over groups of 12 (levels*points), in place (f32).
// ---------------------------------------------------------------------------
__global__ void softmax12_kernel(float* __restrict__ E, long n)
{
  long i = (long)blockIdx.x * blockDim.x + threadIdx.x;
  if (i >= n) return;
  float* e = E + i * 12;
  float v[12];
  float mx = -1e30f;
  #pragma unroll
  for (int j = 0; j < 12; ++j) { v[j] = e[j]; mx = fmaxf(mx, v[j]); }
  float s = 0.0f;
  #pragma unroll
  for (int j = 0; j < 12; ++j) { v[j] = expf(v[j] - mx); s += v[j]; }
  const float inv = 1.0f / s;
  #pragma unroll
  for (int j = 0; j < 12; ++j) e[j] = v[j] * inv;
}

// ---------------------------------------------------------------------------
// Deformable sampling: thread = (b, query, head, d-quarter). 12 bilinear taps
// (zeros padding) over channel-last f32 v; writes f16 into padded chan-last acc.
// ---------------------------------------------------------------------------
__device__ __forceinline__ void corner_acc(const float* __restrict__ vb,
                                           int xi, int yi, float wgt,
                                           float& a0, float& a1, float& a2, float& a3)
{
  const int xc = iclampi(xi, 0, Wc - 1);
  const int yc = iclampi(yi, 0, Hc - 1);
  const float m = (xi >= 0 && xi < Wc && yi >= 0 && yi < Hc) ? wgt : 0.0f;
  const float* v = vb + ((long)yc * Wc + xc) * 64;
  a0 += v[0] * m; a1 += v[1] * m; a2 += v[2] * m; a3 += v[3] * m;
}

__global__ void msdeform_kernel(const float* __restrict__ V,    // (b,3,HW,64) f32
                                const float* __restrict__ Off,  // (b,Lq,96) f32
                                const float* __restrict__ Aw,   // (b,Lq,48) f32
                                _Float16* __restrict__ acc16,   // padded CL, cp=192
                                int B)
{
  const int Lq = 3 * HWc;
  long idx = (long)blockIdx.x * blockDim.x + threadIdx.x;
  if (idx >= (long)B * Lq * 16) return;
  const int q4   = (int)(idx & 3);
  const int head = (int)((idx >> 2) & 3);
  const long t2  = idx >> 4;
  const int r = (int)(t2 % Lq);
  const int b = (int)(t2 / Lq);
  const int tt = r / HWc;
  const int p  = r % HWc;
  const int y = p / Wc, x = p % Wc;

  const float* off = Off + ((long)b * Lq + r) * 96 + head * 24;
  const float* aw  = Aw  + ((long)b * Lq + r) * 48 + head * 12;

  float a0 = 0.0f, a1 = 0.0f, a2 = 0.0f, a3 = 0.0f;
  for (int lvl = 0; lvl < 3; ++lvl) {
    const float* vb = V + (((long)b * 3 + lvl) * HWc) * 64 + head * 16 + q4 * 4;
    #pragma unroll
    for (int pt = 0; pt < 4; ++pt) {
      const float px = (float)x + off[(lvl * 4 + pt) * 2 + 0];
      const float py = (float)y + off[(lvl * 4 + pt) * 2 + 1];
      const float wgt = aw[lvl * 4 + pt];
      const float fx = floorf(px), fy = floorf(py);
      const int ix = (int)fx, iy = (int)fy;
      const float wx = px - fx, wy = py - fy;
      corner_acc(vb, ix,     iy,     (1.0f - wx) * (1.0f - wy) * wgt, a0, a1, a2, a3);
      corner_acc(vb, ix + 1, iy,     wx * (1.0f - wy) * wgt,          a0, a1, a2, a3);
      corner_acc(vb, ix,     iy + 1, (1.0f - wx) * wy * wgt,          a0, a1, a2, a3);
      corner_acc(vb, ix + 1, iy + 1, wx * wy * wgt,                   a0, a1, a2, a3);
    }
  }
  v4h o; o[0] = (_Float16)a0; o[1] = (_Float16)a1; o[2] = (_Float16)a2; o[3] = (_Float16)a3;
  *(v4h*)(acc16 + (((long)b * Hp + y + PADc) * Wp + (x + PADc)) * 192
          + tt * 64 + head * 16 + q4 * 4) = o;
}

__global__ void copy_kernel(const float* __restrict__ src, float* __restrict__ dst, long n)
{
  long i = (long)blockIdx.x * blockDim.x + threadIdx.x;
  const long stride = (long)gridDim.x * blockDim.x;
  for (; i < n; i += stride) dst[i] = src[i];
}

// ---------------------------------------------------------------------------
// Host launcher
// ---------------------------------------------------------------------------
extern "C" void kernel_launch(void* const* d_in, const int* in_sizes, int n_in,
                              void* d_out, int out_size, void* d_ws, size_t ws_size,
                              hipStream_t stream) {
  (void)in_sizes; (void)n_in; (void)out_size; (void)ws_size;

  const float* frame    = (const float*)d_in[0];
  const float* srcframe = (const float*)d_in[1];
  const float* flowf    = (const float*)d_in[2];
  const float* flowb    = (const float*)d_in[3];
  const float* w_qk   = (const float*)d_in[4];   const float* b_qk  = (const float*)d_in[5];
  const float* w_vemb = (const float*)d_in[6];   const float* b_vemb= (const float*)d_in[7];
  const float* w_off  = (const float*)d_in[8];   const float* b_off = (const float*)d_in[9];
  const float* w_att  = (const float*)d_in[10];  const float* b_att = (const float*)d_in[11];
  const float* w_vp   = (const float*)d_in[12];  const float* b_vp  = (const float*)d_in[13];
  const float* w_op   = (const float*)d_in[14];  const float* b_op  = (const float*)d_in[15];
  const float* w_ff   = (const float*)d_in[16];  const float* b_ff  = (const float*)d_in[17];
  const float* w_f1   = (const float*)d_in[18];  const float* b_f1  = (const float*)d_in[19];
  const float* w_f2   = (const float*)d_in[20];  const float* b_f2  = (const float*)d_in[21];

  float* out = (float*)d_out;
  float* ws  = (float*)d_ws;

  const int B = 2, T = 3;
  const int Lq = T * HWc;
  const size_t SA = (size_t)B * T * 64 * HWc;        // 14,155,776 floats

  // ---- workspace layout (float units; all offsets 16B aligned) ----
  const size_t HPW = (size_t)Hp * Wp;                // 38416
  const size_t R1f = B * HPW * 224 / 2;              // qk16 -> attnout16
  const size_t Rhalf = B * HPW * 192 / 2;            // one 2-img 192ch f16 tensor
  float* R1 = ws;                                    // 8,605,184 f
  float* R2 = R1 + R1f;                              // frame16; with R3 => f1in16
  float* R3 = R2 + Rhalf;                            // q16
  float* R4 = R3 + Rhalf;                            // v16 -> acc16 -> f2in16
  float* vproj = R4 + Rhalf;                         // 14,155,776 f
  float* offs  = vproj + SA;                         // 21,233,664 f
  float* attw  = offs + (size_t)B * Lq * 96;         // 10,616,832 f
  float* wreg  = attw + (size_t)B * Lq * 48;

  _Float16* qk16   = (_Float16*)R1;   // cp=224, 2 imgs
  _Float16* att16  = (_Float16*)R1;   // cp=64, 6 imgs (reuse)
  _Float16* frame16= (_Float16*)R2;   // cp=192, 2 imgs
  _Float16* f1in16 = (_Float16*)R2;   // cp=128, 6 imgs (reuse R2+R3)
  _Float16* q16    = (_Float16*)R3;   // cp=192, 2 imgs
  _Float16* v16    = (_Float16*)R4;   // cp=192, 2 imgs
  _Float16* acc16  = (_Float16*)R4;   // cp=192, 2 imgs (reuse)
  _Float16* f2in16 = (_Float16*)R4;   // cp=64, 6 imgs  (reuse)

  // f16 weights, carved from wreg
  _Float16* wh = (_Float16*)wreg;
  _Float16* wqk16   = wh;            wh += (size_t)9 * 192 * 224;
  _Float16* wvemb16 = wh;            wh += (size_t)9 * 192 * 192;
  _Float16* wff16   = wh;            wh += (size_t)9 * 64 * 64;
  _Float16* wf116   = wh;            wh += (size_t)9 * 64 * 128;
  _Float16* wf216   = wh;            wh += (size_t)9 * 64 * 64;
  _Float16* wvpT    = wh;            wh += (size_t)64 * 64;
  _Float16* woffT   = wh;            wh += (size_t)96 * 64;
  _Float16* wattT   = wh;            wh += (size_t)48 * 64;
  _Float16* wopT    = wh;            wh += (size_t)64 * 64;

  auto blocks1d = [](long thr, int bs) { return (int)((thr + bs - 1) / bs); };

  // ---- weight prep ----
  auto convprep = [&](const float* w, _Float16* o, int cout, int cin, int cinp) {
    long thr = (long)9 * cout * cinp;
    convprep_kernel<<<blocks1d(thr, 256), 256, 0, stream>>>(w, o, cout, cin, cinp);
  };
  convprep(w_qk,   wqk16,   192, 200, 224);
  convprep(w_vemb, wvemb16, 192, 192, 192);
  convprep(w_ff,   wff16,   64, 64, 64);
  convprep(w_f1,   wf116,   64, 128, 128);
  convprep(w_f2,   wf216,   64, 64, 64);
  gemmprep_kernel<<<blocks1d(64 * 64, 256), 256, 0, stream>>>(w_vp,  wvpT,  64);
  gemmprep_kernel<<<blocks1d(96 * 64, 256), 256, 0, stream>>>(w_off, woffT, 96);
  gemmprep_kernel<<<blocks1d(48 * 64, 256), 256, 0, stream>>>(w_att, wattT, 48);
  gemmprep_kernel<<<blocks1d(64 * 64, 256), 256, 0, stream>>>(w_op,  wopT,  64);

  auto zero_halo = [&](_Float16* buf, int nimg, int cp) {
    long thr = (long)nimg * HALO * cp;
    zero_halo_kernel<<<blocks1d(thr, 256), 256, 0, stream>>>(buf, nimg, cp);
  };
  auto launch_conv = [&](const _Float16* x16, int cinp, const _Float16* w16,
                         const float* bias, const float* resid, float* o32,
                         _Float16* o16, int o16p, int o16b,
                         int nimg, int cout, int dil, int leaky) {
    long waves = (long)nimg * (cout >> 4) * (HWc >> 4);
    conv3x3_wmma_kernel<<<(int)((waves + 3) / 4), 128, 0, stream>>>(
        x16, cinp, w16, bias, resid, o32, o16, o16p, o16b, nimg, cout, dil, leaky);
  };
  auto launch_gemm = [&](const _Float16* x16, const _Float16* wT, const float* bias,
                         float* o32, long ob, long ot, long op, long on,
                         _Float16* o16, int o16p, int N, int leaky) {
    long waves = (long)B * T * (HWc >> 4) * (N >> 4);
    gemm_wmma_kernel<<<(int)((waves + 3) / 4), 128, 0, stream>>>(
        x16, 192, wT, bias, o32, ob, ot, op, on, o16, o16p, B, N, leaky);
  };

  // ---- 1. qk_in (warp + concat), f16 padded CL cp=224
  zero_halo(qk16, B, 224);
  warp_concat_kernel<<<blocks1d((long)B * HWc, 256), 256, 0, stream>>>(
      frame, flowf, flowb, qk16, B);

  // ---- 2. frame -> f16 padded CL (vemb input)
  zero_halo(frame16, B, 192);
  cvt_cf2cl_kernel<<<blocks1d((long)B * 192 * HWc, 256), 256, 0, stream>>>(
      frame, frame16, B, 192, 192, 0);

  // ---- 3. queries = leaky(conv_qk(qk_in))  -> q16 (cp=192)
  launch_conv(qk16, 224, wqk16, b_qk, nullptr, nullptr, q16, 192, 0, B, 192, 1, 1);
  // ---- 4. value = leaky(conv_vemb(frame))  -> v16 (cp=192)
  launch_conv(frame16, 192, wvemb16, b_vemb, nullptr, nullptr, v16, 192, 0, B, 192, 1, 1);

  // ---- 5. v-proj (f32 chan-last (b,3,HW,64))
  launch_gemm(v16, wvpT, b_vp, vproj, (long)3 * HWc * 64, (long)HWc * 64, 64, 1,
              nullptr, 0, 64, 0);
  // ---- 6. offsets (f32 (b,Lq,96))
  launch_gemm(q16, woffT, b_off, offs, (long)Lq * 96, (long)HWc * 96, 96, 1,
              nullptr, 0, 96, 0);
  // ---- 7. attn logits (f32 (b,Lq,48)) + softmax
  launch_gemm(q16, wattT, b_att, attw, (long)Lq * 48, (long)HWc * 48, 48, 1,
              nullptr, 0, 48, 0);
  softmax12_kernel<<<blocks1d((long)B * Lq * 4, 256), 256, 0, stream>>>(
      attw, (long)B * Lq * 4);

  // ---- 8. deformable sampling -> acc16 (over v16)
  msdeform_kernel<<<blocks1d((long)B * Lq * 16, 256), 256, 0, stream>>>(
      vproj, offs, attw, acc16, B);

  // ---- 9. out-proj -> attnout16 (cp=64, 6 imgs, over qk16 region)
  zero_halo(att16, B * T, 64);
  launch_gemm(acc16, wopT, b_op, nullptr, 0, 0, 0, 0, att16, 64, 64, 0);

  // ---- 10. f1 input buffer: halo + srcframe channels 64..127
  zero_halo(f1in16, B * T, 128);
  cvt_cf2cl_kernel<<<blocks1d((long)B * T * 64 * HWc, 256), 256, 0, stream>>>(
      srcframe, f1in16, B * T, 64, 128, 64);

  // ---- 11. out1 = conv_ff(attn-out) + frame  -> d_out (f32) and f1in16 ch0..63
  launch_conv(att16, 64, wff16, b_ff, frame, out, f1in16, 128, 0, B * T, 64, 1, 0);

  // ---- 12. f1 = leaky(conv_f1(f1in, dil=2)) -> f2in16
  zero_halo(f2in16, B * T, 64);
  launch_conv(f1in16, 128, wf116, b_f1, nullptr, nullptr, f2in16, 64, 0, B * T, 64, 2, 1);

  // ---- 13. d_out = conv_f2(f1) + out1 (in-place residual)
  launch_conv(f2in16, 64, wf216, b_f2, out, out, nullptr, 0, 0, B * T, 64, 1, 0);

  // ---- 14. second output: srcframe
  copy_kernel<<<4096, 256, 0, stream>>>(srcframe, out + SA, (long)SA);
}
